// Node_cls_Module_75788992905515
// MI455X (gfx1250) — matrix operand order repaired and checked by reading, not compiled, not run on hardware
//
#include <hip/hip_runtime.h>
#include <hip/hip_bf16.h>
#include <stdint.h>

// Problem constants (from the reference)
#define N_NODES 50000
#define E_EDGES 500000
#define G_GRAPHS 8
#define IN_F 256
#define HID_F 128
#define OUT_F 64
#define NC_F 10
#define NHEAD 2
#define NPG (N_NODES / G_GRAPHS)
#define NEG_SLOPE 0.2f

typedef __bf16 bf16_t;
typedef bf16_t v16bf __attribute__((ext_vector_type(16)));
typedef bf16_t v8bf  __attribute__((ext_vector_type(8)));
typedef float  v8f   __attribute__((ext_vector_type(8)));

// ---------------------------------------------------------------- helpers ---

__device__ __forceinline__ bf16_t f2bf(float f) {
  union { float f; unsigned u; } v; v.f = f;
  unsigned r = v.u + 0x7FFFu + ((v.u >> 16) & 1u);   // round-to-nearest-even
  unsigned short h = (unsigned short)(r >> 16);
  return __builtin_bit_cast(bf16_t, h);
}

__device__ __forceinline__ void atomicMaxF(float* addr, float val) {
  // sign-aware int punning trick (valid for mixed-sign floats)
  if (val >= 0.0f) atomicMax((int*)addr, __float_as_int(val));
  else             atomicMin((unsigned int*)addr, __float_as_uint(val));
}

__device__ __forceinline__ float lrelu(float x) {
  return x > 0.0f ? x : NEG_SLOPE * x;
}

// ----------------------------------------------------------------- kernels --

__global__ void fill_f32(float* p, float v, long n) {
  long i = (long)blockIdx.x * blockDim.x + threadIdx.x;
  long s = (long)gridDim.x * blockDim.x;
  for (; i < n; i += s) p[i] = v;
}

__global__ void cast_f32_bf16(const float* __restrict__ in, bf16_t* __restrict__ out, long n) {
  long i = (long)blockIdx.x * blockDim.x + threadIdx.x;
  long s = (long)gridDim.x * blockDim.x;
  for (; i < n; i += s) out[i] = f2bf(in[i]);
}

// Cast + rearrange a row-major [K,Nc] f32 weight into WMMA-B fragment order:
//   Bp[((kb*tilesN + tn)*32 + lane)*16 + e]  with lane = g*16 + n,
//   source element = B[kb*32 + g*16 + e, tn*16 + n].
// After packing, a wave's B fragment is one contiguous 32B load per lane.
__global__ void pack_B(const float* __restrict__ B, bf16_t* __restrict__ Bp,
                       int K, int Nc) {
  const int tilesN = Nc >> 4;
  const long total = (long)K * Nc;
  long i = (long)blockIdx.x * blockDim.x + threadIdx.x;
  long stride = (long)gridDim.x * blockDim.x;
  for (; i < total; i += stride) {
    int e = (int)(i & 15);
    long t = i >> 4;
    int lane = (int)(t & 31);
    t >>= 5;
    int tn = (int)(t % tilesN);
    int kb = (int)(t / tilesN);
    int g = lane >> 4, n = lane & 15;
    int k = kb * 32 + g * 16 + e;
    int col = tn * 16 + n;
    Bp[i] = f2bf(B[(long)k * Nc + col]);
  }
}

// per-graph mean of edge_attr (for self-loop fill values)
__global__ void graph_accum(const int* __restrict__ src,
                            const int* __restrict__ batch,
                            const float* __restrict__ ea,
                            float* gsum, float* gcnt) {
  long i = (long)blockIdx.x * blockDim.x + threadIdx.x;
  long s = (long)gridDim.x * blockDim.x;
  for (; i < E_EDGES; i += s) {
    int g = batch[src[i]];
    atomicAdd(&gsum[g], ea[i]);
    atomicAdd(&gcnt[g], 1.0f);
  }
}

__global__ void graph_mean(const float* gsum, const float* gcnt, float* gmean) {
  int g = threadIdx.x;
  if (g < G_GRAPHS) gmean[g] = gsum[g] / gcnt[g];
}

// ce[h] = sum_c We[0, h*C+c] * att_e[h, c]   (edge feature is rank-1)
__global__ void edge_coef(const float* __restrict__ We,
                          const float* __restrict__ att_e,
                          float* ce, int C) {
  int h = threadIdx.x;
  if (h < NHEAD) {
    float s = 0.0f;
    for (int c = 0; c < C; ++c) s += We[h * C + c] * att_e[h * C + c];
    ce[h] = s;
  }
}

// ------------------------------------------------------------ WMMA GEMM -----
// C[M,Nc] = A[M,K] @ B[K,Nc]; A bf16 row-major, B pre-packed (pack_B),
// C f32 row-major. One wave32 per 16x32 C slab (two 16x16 tiles sharing the
// A fragment). K is compile-time -> fully unrolled chain of
// v_wmma_f32_16x16x32_bf16. C layout per CDNA5 ISA 7.12.2; A/B use contiguous k = 16*g + e.
template <int K, int NT>   // NT = Nc/16 (must be even)
__global__ void wmma_gemm_bf16(const bf16_t* __restrict__ A,
                               const bf16_t* __restrict__ Bp,
                               float* __restrict__ C, int M) {
  constexpr int Nc = NT * 16;
  constexpr int pairsN = NT / 2;
  const int pair = blockIdx.x * (blockDim.x >> 5) + (threadIdx.x >> 5);
  const int totalPairs = (M >> 4) * pairsN;
  if (pair >= totalPairs) return;           // wave-uniform branch
  const int tm = pair / pairsN;
  const int tn0 = (pair % pairsN) * 2;

  const int lane = threadIdx.x & 31;
  const int g = lane >> 4;                  // lane group (0/1)
  const int r = lane & 15;                  // row/col within group

  const bf16_t* Arow = A + (size_t)(tm * 16 + r) * K;
  __builtin_prefetch(Arow, 0, 0);           // global_prefetch_b8

  v8f acc0 = {}, acc1 = {};
#pragma unroll
  for (int k0 = 0; k0 < K; k0 += 32) {
    // A fragment: elems 0..15 <-> K = k0+g*16.. (contiguous, matches pack_B)
    v8bf alo = *(const v8bf*)(Arow + k0 + g * 16);
    v8bf ahi = *(const v8bf*)(Arow + k0 + g * 16 + 8);
    v16bf a = __builtin_shufflevector(alo, ahi,
        0, 1, 2, 3, 4, 5, 6, 7, 8, 9, 10, 11, 12, 13, 14, 15);
    // B fragments: contiguous 32B per lane thanks to pack_B
    const bf16_t* bb = Bp + (((size_t)(k0 >> 5) * NT + tn0) * 32 + lane) * 16;
    v16bf b0 = *(const v16bf*)(bb);
    v16bf b1 = *(const v16bf*)(bb + 32 * 16);
    acc0 = __builtin_amdgcn_wmma_f32_16x16x32_bf16(
        false, a, false, b0, (short)0, acc0, false, false);
    acc1 = __builtin_amdgcn_wmma_f32_16x16x32_bf16(
        false, a, false, b1, (short)0, acc1, false, false);
  }
  // C layout: VGPR rr, lane = 16*g + n -> row = rr + 8*g, col = n
#pragma unroll
  for (int rr = 0; rr < 8; ++rr) {
    float* crow = C + (size_t)(tm * 16 + rr + 8 * g) * Nc;
    crow[tn0 * 16 + r]       = acc0[rr];
    crow[(tn0 + 1) * 16 + r] = acc1[rr];
  }
}

// al_s[n,h] = <xp[n,h,:], att_src[h,:]>  (and al_d likewise); one wave per row
__global__ void att_logits(const float* __restrict__ xp,
                           const float* __restrict__ as,
                           const float* __restrict__ ad,
                           float* als, float* ald, int C) {
  int wid = blockIdx.x * (blockDim.x >> 5) + (threadIdx.x >> 5);
  if (wid >= N_NODES * NHEAD) return;
  int lane = threadIdx.x & 31;
  int h = wid % NHEAD;
  const float* row = xp + (long)wid * C;    // [N,H,C] flat: (n*H+h)*C
  float ss = 0.0f, sd = 0.0f;
  for (int c = lane; c < C; c += 32) {
    float v = row[c];
    ss += v * as[h * C + c];
    sd += v * ad[h * C + c];
  }
#pragma unroll
  for (int o = 16; o > 0; o >>= 1) {
    ss += __shfl_xor(ss, o, 32);
    sd += __shfl_xor(sd, o, 32);
  }
  if (lane == 0) { als[wid] = ss; ald[wid] = sd; }
}

// pass 1 of edge softmax: segment max over destinations
__global__ void edge_alpha_max(const int* __restrict__ src,
                               const int* __restrict__ dst,
                               const int* __restrict__ batch,
                               const float* __restrict__ ea,
                               const float* __restrict__ gmean,
                               const float* __restrict__ ce,
                               const float* __restrict__ als,
                               const float* __restrict__ ald,
                               float* maxv) {
  const long total = (long)(E_EDGES + N_NODES) * NHEAD;
  long i = (long)blockIdx.x * blockDim.x + threadIdx.x;
  long stride = (long)gridDim.x * blockDim.x;
  for (; i < total; i += stride) {
    long e = i / NHEAD; int h = (int)(i % NHEAD);
    int s, d; float a;
    if (e < E_EDGES) { s = src[e]; d = dst[e]; a = ea[e]; }
    else { s = d = (int)(e - E_EDGES); a = gmean[batch[s]]; }
    float al = lrelu(als[(long)s * NHEAD + h] + ald[(long)d * NHEAD + h] + a * ce[h]);
    atomicMaxF(&maxv[(long)d * NHEAD + h], al);
  }
}

// pass 2: exp(alpha - max) and segment sum
__global__ void edge_exp_sum(const int* __restrict__ src,
                             const int* __restrict__ dst,
                             const int* __restrict__ batch,
                             const float* __restrict__ ea,
                             const float* __restrict__ gmean,
                             const float* __restrict__ ce,
                             const float* __restrict__ als,
                             const float* __restrict__ ald,
                             const float* __restrict__ maxv,
                             float* ex, float* den) {
  const long total = (long)(E_EDGES + N_NODES) * NHEAD;
  long i = (long)blockIdx.x * blockDim.x + threadIdx.x;
  long stride = (long)gridDim.x * blockDim.x;
  for (; i < total; i += stride) {
    long e = i / NHEAD; int h = (int)(i % NHEAD);
    int s, d; float a;
    if (e < E_EDGES) { s = src[e]; d = dst[e]; a = ea[e]; }
    else { s = d = (int)(e - E_EDGES); a = gmean[batch[s]]; }
    float al = lrelu(als[(long)s * NHEAD + h] + ald[(long)d * NHEAD + h] + a * ce[h]);
    float v = __expf(al - maxv[(long)d * NHEAD + h]);
    ex[i] = v;
    atomicAdd(&den[(long)d * NHEAD + h], v);
  }
}

// pass 3: weighted message scatter: agg[dst,h,c] += xp[src,h,c] * w(e,h)
// float4 per thread: b128 gather + 4 f32 atomics.
__global__ void scatter_msg(const int* __restrict__ src,
                            const int* __restrict__ dst,
                            const float* __restrict__ xp,
                            const float* __restrict__ ex,
                            const float* __restrict__ den,
                            float* agg, int C) {
  const int C2 = NHEAD * C;
  const int Q = C2 >> 2;                     // float4 groups per edge
  const long total = (long)(E_EDGES + N_NODES) * Q;
  long i = (long)blockIdx.x * blockDim.x + threadIdx.x;
  long stride = (long)gridDim.x * blockDim.x;
  for (; i < total; i += stride) {
    long e = i / Q; int q = (int)(i % Q);
    int hc = q * 4; int h = hc / C;          // 4 channels stay in one head
    int s, d;
    if (e < E_EDGES) { s = src[e]; d = dst[e]; }
    else { s = d = (int)(e - E_EDGES); }
    float w = ex[e * NHEAD + h] / (den[(long)d * NHEAD + h] + 1e-16f);
    const float4 v = *((const float4*)(xp + (long)s * C2) + q);
    float* ag = agg + (long)d * C2 + hc;
    atomicAdd(ag + 0, v.x * w);
    atomicAdd(ag + 1, v.y * w);
    atomicAdd(ag + 2, v.z * w);
    atomicAdd(ag + 3, v.w * w);
  }
}

// out[n,c] = (agg[n,0,c] + agg[n,1,c]) / 2 + b[c]  (optional relu)
__global__ void head_mean_bias(const float* __restrict__ agg,
                               const float* __restrict__ b,
                               float* out, int C, int do_relu) {
  const long total = (long)N_NODES * C;
  long i = (long)blockIdx.x * blockDim.x + threadIdx.x;
  long stride = (long)gridDim.x * blockDim.x;
  for (; i < total; i += stride) {
    long n = i / C; int c = (int)(i % C);
    float v = 0.5f * (agg[n * 2 * C + c] + agg[n * 2 * C + C + c]) + b[c];
    if (do_relu) v = fmaxf(v, 0.0f);
    out[i] = v;
  }
}

// pred[n,k] = h2[n,:] @ Wc[:,k] + bc[k]   (K=64, NC=10 -> plain VALU is fine)
__global__ void classifier_k(const float* __restrict__ h2,
                             const float* __restrict__ Wc,
                             const float* __restrict__ bc,
                             float* out) {
  const long total = (long)N_NODES * NC_F;
  long i = (long)blockIdx.x * blockDim.x + threadIdx.x;
  long stride = (long)gridDim.x * blockDim.x;
  for (; i < total; i += stride) {
    long n = i / NC_F; int k = (int)(i % NC_F);
    float s = bc[k];
#pragma unroll 8
    for (int c = 0; c < OUT_F; ++c) s += h2[n * OUT_F + c] * Wc[c * NC_F + k];
    out[i] = s;
  }
}

__global__ void y_copy(const int* __restrict__ y, float* out) {
  long i = (long)blockIdx.x * blockDim.x + threadIdx.x;
  long stride = (long)gridDim.x * blockDim.x;
  for (; i < N_NODES; i += stride) out[i] = (float)y[i];
}

// ------------------------------------------------------------------ host ----

extern "C" void kernel_launch(void* const* d_in, const int* in_sizes, int n_in,
                              void* d_out, int out_size, void* d_ws, size_t ws_size,
                              hipStream_t stream) {
  const float* x      = (const float*)d_in[0];
  const int*   ei     = (const int*)d_in[1];
  const int*   src    = ei;
  const int*   dst    = ei + E_EDGES;
  const float* eattr  = (const float*)d_in[2];
  const int*   batch  = (const int*)d_in[3];
  const int*   y      = (const int*)d_in[4];
  const float* W1     = (const float*)d_in[5];
  const float* as1    = (const float*)d_in[6];
  const float* ad1    = (const float*)d_in[7];
  const float* We1    = (const float*)d_in[8];
  const float* ae1    = (const float*)d_in[9];
  const float* b1     = (const float*)d_in[10];
  const float* W2     = (const float*)d_in[11];
  const float* as2    = (const float*)d_in[12];
  const float* ad2    = (const float*)d_in[13];
  const float* We2    = (const float*)d_in[14];
  const float* ae2    = (const float*)d_in[15];
  const float* b2     = (const float*)d_in[16];
  const float* Wc     = (const float*)d_in[17];
  const float* bc     = (const float*)d_in[18];

  char* ws = (char*)d_ws;
  size_t off = 0;
  auto alloc = [&](size_t bytes) {
    size_t o = off;
    off += (bytes + 255) & ~(size_t)255;
    return o;
  };

  const long ET = (long)E_EDGES + N_NODES;

  float*  gsum  = (float*)(ws + alloc(G_GRAPHS * 4));
  float*  gcnt  = (float*)(ws + alloc(G_GRAPHS * 4));
  float*  gmean = (float*)(ws + alloc(G_GRAPHS * 4));
  float*  ce1   = (float*)(ws + alloc(NHEAD * 4));
  float*  ce2   = (float*)(ws + alloc(NHEAD * 4));
  bf16_t* xb    = (bf16_t*)(ws + alloc((size_t)N_NODES * IN_F * 2));
  bf16_t* W1p   = (bf16_t*)(ws + alloc((size_t)IN_F * NHEAD * HID_F * 2));
  float*  xp1   = (float*)(ws + alloc((size_t)N_NODES * NHEAD * HID_F * 4));
  float*  als1  = (float*)(ws + alloc((size_t)N_NODES * NHEAD * 4));
  float*  ald1  = (float*)(ws + alloc((size_t)N_NODES * NHEAD * 4));
  float*  maxv1 = (float*)(ws + alloc((size_t)N_NODES * NHEAD * 4));
  float*  den1  = (float*)(ws + alloc((size_t)N_NODES * NHEAD * 4));
  float*  ex1   = (float*)(ws + alloc((size_t)ET * NHEAD * 4));
  float*  agg1  = (float*)(ws + alloc((size_t)N_NODES * NHEAD * HID_F * 4));
  float*  h1    = (float*)(ws + alloc((size_t)N_NODES * HID_F * 4));
  bf16_t* h1b   = (bf16_t*)(ws + alloc((size_t)N_NODES * HID_F * 2));
  bf16_t* W2p   = (bf16_t*)(ws + alloc((size_t)HID_F * NHEAD * OUT_F * 2));
  float*  xp2   = (float*)(ws + alloc((size_t)N_NODES * NHEAD * OUT_F * 4));
  float*  als2  = (float*)(ws + alloc((size_t)N_NODES * NHEAD * 4));
  float*  ald2  = (float*)(ws + alloc((size_t)N_NODES * NHEAD * 4));
  float*  maxv2 = (float*)(ws + alloc((size_t)N_NODES * NHEAD * 4));
  float*  den2  = (float*)(ws + alloc((size_t)N_NODES * NHEAD * 4));
  float*  ex2   = (float*)(ws + alloc((size_t)ET * NHEAD * 4));
  float*  agg2  = (float*)(ws + alloc((size_t)N_NODES * NHEAD * OUT_F * 4));
  float*  h2    = (float*)(ws + alloc((size_t)N_NODES * OUT_F * 4));

  // ---- self-loop edge-attr fill values -------------------------------------
  hipMemsetAsync(gsum, 0, G_GRAPHS * 4, stream);
  hipMemsetAsync(gcnt, 0, G_GRAPHS * 4, stream);
  graph_accum<<<2048, 256, 0, stream>>>(src, batch, eattr, gsum, gcnt);
  graph_mean<<<1, 32, 0, stream>>>(gsum, gcnt, gmean);
  edge_coef<<<1, 32, 0, stream>>>(We1, ae1, ce1, HID_F);
  edge_coef<<<1, 32, 0, stream>>>(We2, ae2, ce2, OUT_F);

  // ---- layer 1 -------------------------------------------------------------
  cast_f32_bf16<<<2048, 256, 0, stream>>>(x, xb, (long)N_NODES * IN_F);
  pack_B<<<256, 256, 0, stream>>>(W1, W1p, IN_F, NHEAD * HID_F);
  // 16x32 slabs: 3125 * 8 = 25000 waves; 4 waves/block -> 6250 blocks
  wmma_gemm_bf16<IN_F, 16><<<6250, 128, 0, stream>>>(xb, W1p, xp1, N_NODES);
  att_logits<<<12500, 256, 0, stream>>>(xp1, as1, ad1, als1, ald1, HID_F);

  fill_f32<<<512, 256, 0, stream>>>(maxv1, -1e30f, (long)N_NODES * NHEAD);
  hipMemsetAsync(den1, 0, (size_t)N_NODES * NHEAD * 4, stream);
  hipMemsetAsync(agg1, 0, (size_t)N_NODES * NHEAD * HID_F * 4, stream);

  edge_alpha_max<<<2048, 256, 0, stream>>>(src, dst, batch, eattr, gmean, ce1,
                                           als1, ald1, maxv1);
  edge_exp_sum<<<2048, 256, 0, stream>>>(src, dst, batch, eattr, gmean, ce1,
                                         als1, ald1, maxv1, ex1, den1);
  scatter_msg<<<4096, 256, 0, stream>>>(src, dst, xp1, ex1, den1, agg1, HID_F);
  head_mean_bias<<<2048, 256, 0, stream>>>(agg1, b1, h1, HID_F, /*relu=*/1);

  // ---- layer 2 -------------------------------------------------------------
  cast_f32_bf16<<<2048, 256, 0, stream>>>(h1, h1b, (long)N_NODES * HID_F);
  pack_B<<<64, 256, 0, stream>>>(W2, W2p, HID_F, NHEAD * OUT_F);
  // 16x32 slabs: 3125 * 4 = 12500 waves; 4 waves/block -> 3125 blocks
  wmma_gemm_bf16<HID_F, 8><<<3125, 128, 0, stream>>>(h1b, W2p, xp2, N_NODES);
  att_logits<<<12500, 256, 0, stream>>>(xp2, as2, ad2, als2, ald2, OUT_F);

  fill_f32<<<512, 256, 0, stream>>>(maxv2, -1e30f, (long)N_NODES * NHEAD);
  hipMemsetAsync(den2, 0, (size_t)N_NODES * NHEAD * 4, stream);
  hipMemsetAsync(agg2, 0, (size_t)N_NODES * NHEAD * OUT_F * 4, stream);

  edge_alpha_max<<<2048, 256, 0, stream>>>(src, dst, batch, eattr, gmean, ce2,
                                           als2, ald2, maxv2);
  edge_exp_sum<<<2048, 256, 0, stream>>>(src, dst, batch, eattr, gmean, ce2,
                                         als2, ald2, maxv2, ex2, den2);
  scatter_msg<<<4096, 256, 0, stream>>>(src, dst, xp2, ex2, den2, agg2, OUT_F);
  head_mean_bias<<<2048, 256, 0, stream>>>(agg2, b2, h2, OUT_F, /*relu=*/0);

  // ---- classifier + outputs ------------------------------------------------
  float* pred = (float*)d_out;
  classifier_k<<<2048, 256, 0, stream>>>(h2, Wc, bc, pred);
  y_copy<<<256, 256, 0, stream>>>(y, pred + (size_t)N_NODES * NC_F);
}
